// Loss_67224828117153
// MI455X (gfx1250) — compile-verified, matrix operand order checked by reading, not run on hardware
//
#include <hip/hip_runtime.h>
#include <hip/hip_bf16.h>

#define NB 16
#define IH 384
#define IW 768
#define NP 24576          // partial slots, multiple of 512
#define TS 16
#define TH2 18            // TS + 2 halo
#define TA (TH2*TH2)      // 324

typedef __attribute__((ext_vector_type(2)))  float    v2f;
typedef __attribute__((ext_vector_type(8)))  float    v8f;
typedef __attribute__((ext_vector_type(16))) _Float16 v16h;

#define SSIM_C1 (0.01f*0.01f)
#define SSIM_C2 (0.03f*0.03f)

__global__ __launch_bounds__(256) void zero_kernel(float* __restrict__ p, int n) {
  int i = blockIdx.x * blockDim.x + threadIdx.x;
  if (i < n) p[i] = 0.0f;
}

// Align-corners bilinear resize of both full-res images to (h, w).
__global__ __launch_bounds__(256) void resize_kernel(const float* __restrict__ srcL,
                                                     const float* __restrict__ srcR,
                                                     float* __restrict__ dstL,
                                                     float* __restrict__ dstR,
                                                     int h, int w) {
  long long total = 2LL * NB * 3 * h * w;
  long long idx = (long long)blockIdx.x * blockDim.x + threadIdx.x;
  if (idx >= total) return;
  long long per = (long long)NB * 3 * h * w;
  int which = (int)(idx / per);
  long long rem = idx % per;
  int x = (int)(rem % w);
  int y = (int)((rem / w) % h);
  int c = (int)((rem / ((long long)w * h)) % 3);
  int b = (int)(rem / ((long long)w * h * 3));

  float sy = (float)y * (float)(IH - 1) / (float)(h - 1);
  float sx = (float)x * (float)(IW - 1) / (float)(w - 1);
  int y0 = (int)floorf(sy); float wy = sy - (float)y0; int y1 = min(y0 + 1, IH - 1);
  int x0 = (int)floorf(sx); float wx = sx - (float)x0; int x1 = min(x0 + 1, IW - 1);

  const float* src = which ? srcR : srcL;
  const float* p0 = src + ((long long)(b * 3 + c) * IH + y0) * IW;
  const float* p1 = src + ((long long)(b * 3 + c) * IH + y1) * IW;
  float r0 = p0[x0] * (1.0f - wy) + p1[x0] * wy;
  float r1 = p0[x1] * (1.0f - wy) + p1[x1] * wy;
  float v  = r0 * (1.0f - wx) + r1 * wx;

  float* dst = which ? dstR : dstL;
  dst[((long long)(b * 3 + c) * h + y) * w + x] = v;
}

__device__ __forceinline__ float gather1(const float* __restrict__ row, float px, int w) {
  float xf = floorf(px); float f = px - xf;
  int i0 = (int)xf, i1 = i0 + 1;
  float w0 = (i0 >= 0 && i0 < w) ? (1.0f - f) : 0.0f;
  float w1 = (i1 >= 0 && i1 < w) ? f : 0.0f;
  int c0 = min(max(i0, 0), w - 1), c1 = min(max(i1, 0), w - 1);
  return row[c0] * w0 + row[c1] * w1;
}

__device__ __forceinline__ float ssim_win(const float* __restrict__ X,
                                          const float* __restrict__ Y,
                                          int ly, int lx) {
  float sx = 0.f, sy = 0.f, sxx = 0.f, syy = 0.f, sxy = 0.f;
#pragma unroll
  for (int dy = -1; dy <= 1; ++dy)
#pragma unroll
    for (int dx = -1; dx <= 1; ++dx) {
      float xv = X[(ly + dy) * TH2 + (lx + dx)];
      float yv = Y[(ly + dy) * TH2 + (lx + dx)];
      sx += xv; sy += yv;
      sxx = fmaf(xv, xv, sxx);
      syy = fmaf(yv, yv, syy);
      sxy = fmaf(xv, yv, sxy);
    }
  const float inv9 = 1.0f / 9.0f;
  float mx = sx * inv9, my = sy * inv9;
  float vx = sxx * inv9 - mx * mx;
  float vy = syy * inv9 - my * my;
  float cc = sxy * inv9 - mx * my;
  float s = ((2.0f * mx * my + SSIM_C1) * (2.0f * cc + SSIM_C2)) /
            ((mx * mx + my * my + SSIM_C1) * (vx + vy + SSIM_C2));
  return fminf(fmaxf((1.0f - s) * 0.5f, 0.0f), 1.0f);
}

// Fused per-level kernel: warp + L1 + SSIM + LR-consistency + smoothness.
__global__ __launch_bounds__(256) void level_kernel(
    const float* __restrict__ lpy, const float* __restrict__ rpy,
    const float* __restrict__ disp, int h, int w,
    float smw, float* __restrict__ partials, int pbase) {
  __shared__ float s_dl[TA], s_dr[TA];
  __shared__ float s_lpy[3][TA], s_rpy[3][TA];
  __shared__ float s_el[3][TA], s_er[3][TA];
  __shared__ float red[5][8];

  const int tid = threadIdx.y * TS + threadIdx.x;
  const int b = blockIdx.z;
  const int x0g = blockIdx.x * TS, y0g = blockIdx.y * TS;
  const float wm1 = (float)(w - 1);
  const size_t hw = (size_t)h * w;
  const float* dl_g = disp + (size_t)b * 2 * hw;
  const float* dr_g = dl_g + hw;
  const float* lb = lpy + (size_t)b * 3 * hw;
  const float* rb = rpy + (size_t)b * 3 * hw;

  // Pass A: stage disparities and both pyramid images (tile + 1 halo, clamped).
  for (int i = tid; i < TA; i += TS * TS) {
    int ly = i / TH2, lx = i % TH2;
    int gy = min(max(y0g + ly - 1, 0), h - 1);
    int gx = min(max(x0g + lx - 1, 0), w - 1);
    size_t o = (size_t)gy * w + gx;
    s_dl[i] = dl_g[o];
    s_dr[i] = dr_g[o];
#pragma unroll
    for (int c = 0; c < 3; ++c) {
      s_lpy[c][i] = lb[c * hw + o];
      s_rpy[c][i] = rb[c * hw + o];
    }
  }
  __syncthreads();

  // Pass B: warped estimates into LDS (gathers hit L2).
  for (int i = tid; i < TA; i += TS * TS) {
    int ly = i / TH2, lx = i % TH2;
    int gy = min(max(y0g + ly - 1, 0), h - 1);
    int gx = min(max(x0g + lx - 1, 0), w - 1);
    size_t ro = (size_t)gy * w;
    {
      float px = (float)gx - s_dl[i] * wm1;           // left_est <- warp(rpy, dl, -1)
      float xf = floorf(px); float f = px - xf;
      int i0 = (int)xf, i1 = i0 + 1;
      float w0 = (i0 >= 0 && i0 < w) ? (1.0f - f) : 0.0f;
      float w1 = (i1 >= 0 && i1 < w) ? f : 0.0f;
      int c0 = min(max(i0, 0), w - 1), c1 = min(max(i1, 0), w - 1);
#pragma unroll
      for (int c = 0; c < 3; ++c)
        s_el[c][i] = rb[c * hw + ro + c0] * w0 + rb[c * hw + ro + c1] * w1;
    }
    {
      float px = (float)gx + s_dr[i] * wm1;           // right_est <- warp(lpy, dr, +1)
      float xf = floorf(px); float f = px - xf;
      int i0 = (int)xf, i1 = i0 + 1;
      float w0 = (i0 >= 0 && i0 < w) ? (1.0f - f) : 0.0f;
      float w1 = (i1 >= 0 && i1 < w) ? f : 0.0f;
      int c0 = min(max(i0, 0), w - 1), c1 = min(max(i1, 0), w - 1);
#pragma unroll
      for (int c = 0; c < 3; ++c)
        s_er[c][i] = lb[c * hw + ro + c0] * w0 + lb[c * hw + ro + c1] * w1;
    }
  }
  __syncthreads();

  // Accumulation: every thread owns one pixel.
  float a_l1 = 0.f, a_ss = 0.f, a_lr = 0.f, a_smx = 0.f, a_smy = 0.f;
  const int gx = x0g + threadIdx.x, gy = y0g + threadIdx.y;
  if (gx < w && gy < h) {
    const int li = (threadIdx.y + 1) * TH2 + (threadIdx.x + 1);
    const float dlv = s_dl[li], drv = s_dr[li];
#pragma unroll
    for (int c = 0; c < 3; ++c)
      a_l1 += fabsf(s_el[c][li] - s_lpy[c][li]) + fabsf(s_er[c][li] - s_rpy[c][li]);

    // LR consistency: warp the opposite disparity map and compare.
    {
      const float* rowdr = dr_g + (size_t)gy * w;
      const float* rowdl = dl_g + (size_t)gy * w;
      float rl = gather1(rowdr, (float)gx - dlv * wm1, w);
      float lr = gather1(rowdl, (float)gx + drv * wm1, w);
      a_lr += fabsf(rl - dlv) + fabsf(lr - drv);
    }
    // Edge-aware smoothness.
    if (gx + 1 < w) {
      float sL = 0.f, sR = 0.f;
#pragma unroll
      for (int c = 0; c < 3; ++c) {
        sL += fabsf(s_lpy[c][li] - s_lpy[c][li + 1]);
        sR += fabsf(s_rpy[c][li] - s_rpy[c][li + 1]);
      }
      a_smx += fabsf(dlv - s_dl[li + 1]) * expf(-sL * (1.0f / 3.0f));
      a_smx += fabsf(drv - s_dr[li + 1]) * expf(-sR * (1.0f / 3.0f));
    }
    if (gy + 1 < h) {
      float sL = 0.f, sR = 0.f;
#pragma unroll
      for (int c = 0; c < 3; ++c) {
        sL += fabsf(s_lpy[c][li] - s_lpy[c][li + TH2]);
        sR += fabsf(s_rpy[c][li] - s_rpy[c][li + TH2]);
      }
      a_smy += fabsf(dlv - s_dl[li + TH2]) * expf(-sL * (1.0f / 3.0f));
      a_smy += fabsf(drv - s_dr[li + TH2]) * expf(-sR * (1.0f / 3.0f));
    }
    // SSIM, VALID 3x3 windows only.
    if (gx >= 1 && gx <= w - 2 && gy >= 1 && gy <= h - 2) {
#pragma unroll
      for (int c = 0; c < 3; ++c) {
        a_ss += ssim_win(s_el[c], s_lpy[c], threadIdx.y + 1, threadIdx.x + 1);
        a_ss += ssim_win(s_er[c], s_rpy[c], threadIdx.y + 1, threadIdx.x + 1);
      }
    }
  }

  // wave32 shuffle reduce, then cross-wave via LDS.
  float acc[5] = {a_l1, a_ss, a_lr, a_smx, a_smy};
#pragma unroll
  for (int k = 0; k < 5; ++k) {
    float v = acc[k];
#pragma unroll
    for (int off = 16; off > 0; off >>= 1) v += __shfl_down(v, off, 32);
    acc[k] = v;
  }
  const int wave = tid >> 5, lane = tid & 31;
  if (lane == 0) {
#pragma unroll
    for (int k = 0; k < 5; ++k) red[k][wave] = acc[k];
  }
  __syncthreads();
  if (tid == 0) {
    float s[5];
#pragma unroll
    for (int k = 0; k < 5; ++k) {
      float v = 0.f;
#pragma unroll
      for (int j = 0; j < 8; ++j) v += red[k][j];
      s[k] = v;
    }
    float inv_l1  = 1.0f / ((float)NB * 3.0f * (float)h * (float)w);
    float inv_ss  = 1.0f / ((float)NB * 3.0f * (float)(h - 2) * (float)(w - 2));
    float inv_lr  = 1.0f / ((float)NB * (float)h * (float)w);
    float inv_smx = 1.0f / ((float)NB * (float)h * (float)(w - 1));
    float inv_smy = 1.0f / ((float)NB * (float)(h - 1) * (float)w);
    float partial = 0.5f * s[0] * inv_l1 + 0.5f * s[1] * inv_ss + s[2] * inv_lr +
                    smw * (s[3] * inv_smx + s[4] * inv_smy);
    int id = blockIdx.x + gridDim.x * (blockIdx.y + gridDim.y * blockIdx.z);
    partials[pbase + id] = partial;   // one slot per block: deterministic, no atomics
  }
}

// Final reduction of NP partials on one wave via the matrix pipe:
// D = A x ones + C sums 64 f32 values per V_WMMA_F32_16X16X4_F32.
__global__ void reduce_kernel(const float* __restrict__ P, float* __restrict__ out) {
  int lane = threadIdx.x;
  float total;
#if defined(__HIP_DEVICE_COMPILE__) && __has_builtin(__builtin_amdgcn_wmma_f32_16x16x4_f32)
  v8f c = {0.f, 0.f, 0.f, 0.f, 0.f, 0.f, 0.f, 0.f};
  v2f bo = {1.0f, 1.0f};
  for (int t = 0; t < NP; t += 64) {
    v2f a = {P[t + lane * 2], P[t + lane * 2 + 1]};
    c = __builtin_amdgcn_wmma_f32_16x16x4_f32(false, a, false, bo, (short)0, c, false, false);
  }
  float v = c[0] + c[1] + c[2] + c[3] + c[4] + c[5] + c[6] + c[7];
  v += __shfl_down(v, 16, 32);   // rows 0-7 (lane n) + rows 8-15 (lane n+16), col n
  total = v;
#elif defined(__HIP_DEVICE_COMPILE__)
  v8f c = {0.f, 0.f, 0.f, 0.f, 0.f, 0.f, 0.f, 0.f};
  v16h bo;
#pragma unroll
  for (int j = 0; j < 16; ++j) bo[j] = (_Float16)1.0f;
  for (int t = 0; t < NP; t += 512) {
    v16h a;
#pragma unroll
    for (int j = 0; j < 16; ++j) a[j] = (_Float16)P[t + lane * 16 + j];
    c = __builtin_amdgcn_wmma_f32_16x16x32_f16(false, a, false, bo, (short)0, c, false, false);
  }
  float v = c[0] + c[1] + c[2] + c[3] + c[4] + c[5] + c[6] + c[7];
  v += __shfl_down(v, 16, 32);
  total = v;
#else
  float v = 0.f;
  for (int t = lane; t < NP; t += 32) v += P[t];
  for (int off = 16; off > 0; off >>= 1) v += __shfl_down(v, off, 32);
  total = v;
#endif
  if (lane == 0) out[0] = total;
}

extern "C" void kernel_launch(void* const* d_in, const int* in_sizes, int n_in,
                              void* d_out, int out_size, void* d_ws, size_t ws_size,
                              hipStream_t stream) {
  (void)in_sizes; (void)n_in; (void)out_size; (void)ws_size;
  const float* disp[4] = {(const float*)d_in[0], (const float*)d_in[1],
                          (const float*)d_in[2], (const float*)d_in[3]};
  const float* left  = (const float*)d_in[4];
  const float* right = (const float*)d_in[5];
  float* ws = (float*)d_ws;
  float* parts = ws;
  size_t off = NP;
  const int hs[4] = {384, 192, 96, 48};
  const int wsv[4] = {768, 384, 192, 96};
  float* lp[4] = {nullptr, nullptr, nullptr, nullptr};
  float* rp[4] = {nullptr, nullptr, nullptr, nullptr};
  for (int i = 1; i < 4; ++i) {
    size_t n = (size_t)NB * 3 * hs[i] * wsv[i];
    lp[i] = ws + off; off += n;
    rp[i] = ws + off; off += n;
  }

  zero_kernel<<<(NP + 255) / 256, 256, 0, stream>>>(parts, NP);

  for (int i = 1; i < 4; ++i) {
    long long total = 2LL * NB * 3 * hs[i] * wsv[i];
    int nblk = (int)((total + 255) / 256);
    resize_kernel<<<nblk, 256, 0, stream>>>(left, right, lp[i], rp[i], hs[i], wsv[i]);
  }

  int pbase = 0;
  for (int i = 0; i < 4; ++i) {
    int h = hs[i], w = wsv[i];
    dim3 grid(w / TS, h / TS, NB);   // all level dims divide by 16
    const float* L = (i == 0) ? left  : lp[i];
    const float* R = (i == 0) ? right : rp[i];
    float smw = 0.1f / (float)(1 << i);
    level_kernel<<<grid, dim3(TS, TS), 0, stream>>>(L, R, disp[i], h, w, smw, parts, pbase);
    pbase += (int)(grid.x * grid.y * grid.z);
  }

  reduce_kernel<<<1, 32, 0, stream>>>(parts, (float*)d_out);
}
